// Decoder_19705309954222
// MI455X (gfx1250) — matrix operand order repaired and checked by reading, not compiled, not run on hardware
//
#include <hip/hip_runtime.h>
#include <hip/hip_bf16.h>
#include <stdint.h>

// ---------------------------------------------------------------------------
// LSTM (Keras gate order i,f,g,o), B=32768, T=7, F=36, U=256.
// One fused recurrent GEMM per step:
//   z = [x_t | h] (32x320 bf16, K padded)  @  [Wk;Wr] (320x1024 bf16)  + b
// via v_wmma_f32_16x16x32_bf16, f32 accumulation, gates evaluated in regs.
// BM=32: each B fragment (from L2) is reused by two M-tiles, halving the
// dominant L2 weight traffic. Scalar-offset laundering per timestep blocks
// LICM from hoisting weight loads; kc-loop unroll capped at 1 bounds the
// load-pipelining register pressure (target < 256 VGPRs, no spills).
// ---------------------------------------------------------------------------

typedef unsigned short u16;
typedef __bf16 bf16_t;
typedef bf16_t v16bf __attribute__((ext_vector_type(16)));
typedef float  v8f   __attribute__((ext_vector_type(8)));
typedef u16    v16u  __attribute__((ext_vector_type(16)));
typedef u16    v8u   __attribute__((ext_vector_type(8)));

#define LSTM_T   7
#define LSTM_F   36
#define LSTM_U   256
#define LSTM_G4U 1024
#define KPAD     320      // 36 + 256 padded to 10 chunks of K=32
#define NKC      10       // K chunks
#define NNT      64       // N tiles of 16 (1024/16)
#define BM       32       // rows per block (2 M-tiles, B-fragment reuse x2)

__device__ __forceinline__ u16 f2bf(float f) {
  unsigned u = __builtin_bit_cast(unsigned, f);
  u += 0x7fffu + ((u >> 16) & 1u);   // round-to-nearest-even
  return (u16)(u >> 16);
}

// Fast tanh: V_TANH_F32 when the builtin exists, else exp-based.
__device__ __forceinline__ float fast_tanh(float x) {
#if __has_builtin(__builtin_amdgcn_tanhf)
  return __builtin_amdgcn_tanhf(x);
#else
  // tanh(x) = 1 - 2/(exp(2x)+1)
  return 1.0f - 2.0f / (__expf(2.0f * x) + 1.0f);
#endif
}

__device__ __forceinline__ float sigm(float x) {
  return 0.5f * (1.0f + fast_tanh(0.5f * x));
}

// ---------------------------------------------------------------------------
// Pack [Wk;Wr] (fp32) into bf16 B-fragment-major layout in workspace.
// Packed element index: idx = ((nt*NKC + kc) * 512) + lane*16 + e
//   lane 0..15  -> K = kc*32 + 16*(e/8) + (e%8)     (e = 0..15), col N = nt*16 + lane
//   lane 16..31 -> K = kc*32 + 16*(e/8) + 8 + (e%8),              col N = nt*16 + lane-16
// Matches the V_WMMA 16-bit B-matrix (32x16) per-lane VGPR layout, so the
// main kernel loads one contiguous 32B chunk per lane per fragment.
// ---------------------------------------------------------------------------
__global__ void lstm_pack_weights(const float* __restrict__ Wk,
                                  const float* __restrict__ Wr,
                                  u16* __restrict__ wsB) {
  int idx = blockIdx.x * blockDim.x + threadIdx.x;
  const int total = NNT * NKC * 512;
  if (idx >= total) return;
  int e    = idx & 15;
  int lane = (idx >> 4) & 31;
  int blk  = idx >> 9;
  int kc   = blk % NKC;
  int nt   = blk / NKC;
  int K = kc * 32 + ((e >> 3) << 4) + ((lane >> 4) << 3) + (e & 7);
  int N = nt * 16 + (lane & 15);
  float v = 0.0f;
  if (K < LSTM_F)               v = Wk[K * LSTM_G4U + N];
  else if (K < LSTM_F + LSTM_U) v = Wr[(K - LSTM_F) * LSTM_G4U + N];
  wsB[idx] = f2bf(v);
}

// ---------------------------------------------------------------------------
// Main fused LSTM kernel. 256 threads = 8 waves (wave32). Block owns BM=32
// rows for all 7 steps. Wave w owns units [w*32, w*32+32) for BOTH M-tiles:
// 2(m) x 4(gate) x 2(j) = 16 f32 C-tiles (128 VGPRs); each B fragment is
// loaded once and used by two WMMAs (both M-tiles).
// ---------------------------------------------------------------------------
__global__ void __launch_bounds__(256)
lstm_main(const float* __restrict__ x,
          const float* __restrict__ h0,
          const float* __restrict__ c0,
          const float* __restrict__ b,
          const u16*   __restrict__ wsB,
          float*       __restrict__ out) {
  // Row-major bf16 A tile: cols [0,36)=x_t, [36,292)=h, [292,320)=zero pad.
  __shared__ u16 A[BM][KPAD];   // 20 KB

  const int tid  = threadIdx.x;
  const int w    = tid >> 5;      // wave id 0..7
  const int lane = tid & 31;
  const int lh   = lane >> 4;     // lane-half (selects K sub-range / row+8)
  const int ll   = lane & 15;
  const int row0 = blockIdx.x * BM;

  // Zero pad columns [292,320): 32*28 = 896 elements
  for (int i = tid; i < BM * (KPAD - LSTM_F - LSTM_U); i += 256) {
    A[i / 28][LSTM_F + LSTM_U + (i % 28)] = 0;
  }
  // h0 -> LDS (bf16): 32*256 = 8192 elements
  for (int i = tid; i < BM * LSTM_U; i += 256) {
    int r = i >> 8, u = i & 255;
    A[r][LSTM_F + u] = f2bf(h0[(row0 + r) * LSTM_U + u]);
  }

  // Per-lane bias (depends on column only; broadcast across C-tile rows)
  float bias[4][2];
#pragma unroll
  for (int g = 0; g < 4; ++g)
#pragma unroll
    for (int j = 0; j < 2; ++j)
      bias[g][j] = b[g * LSTM_U + w * 32 + j * 16 + ll];

  // c state in registers, C-tile layout: elem r, lane -> row r+8*lh, col ll
  v8f cst[2][2];
#pragma unroll
  for (int m = 0; m < 2; ++m)
#pragma unroll
    for (int j = 0; j < 2; ++j)
#pragma unroll
      for (int r = 0; r < 8; ++r)
        cst[m][j][r] = c0[(row0 + m * 16 + r + 8 * lh) * LSTM_U +
                          w * 32 + j * 16 + ll];

#pragma unroll 1
  for (int t = 0; t < LSTM_T; ++t) {
    // Launder a scalar offset (stays 0): makes weight-load addresses loop-
    // variant so LICM cannot hoist the B-fragment loads out of the t-loop,
    // while keeping addressing rooted at the global kernarg pointer
    // (global_load_b128, not flat_load, no scratch spills).
    int toff = 0;
    asm volatile("" : "+s"(toff));
    const u16* wp = wsB + toff;

    // Stage x_t (fp32 -> bf16) into A cols [0,36): 32*36 = 1152 elements
    for (int i = tid; i < BM * LSTM_F; i += 256) {
      int r = i / LSTM_F, f = i % LSTM_F;
      A[r][f] = f2bf(x[(row0 + r) * (LSTM_T * LSTM_F) + t * LSTM_F + f]);
    }
    __syncthreads();   // x_t + h_t visible to all waves

    // z accumulators, seeded with bias
    v8f acc[2][4][2];
#pragma unroll
    for (int m = 0; m < 2; ++m)
#pragma unroll
      for (int g = 0; g < 4; ++g)
#pragma unroll
        for (int j = 0; j < 2; ++j)
#pragma unroll
          for (int r = 0; r < 8; ++r)
            acc[m][g][j][r] = bias[g][j];

    // GEMM: K = 320 in 10 chunks of 32. kc rolled (#pragma unroll 1) to
    // bound in-flight fragments; B loaded once, used for both M-tiles.
#pragma unroll 1
    for (int kc = 0; kc < NKC; ++kc) {
      // A fragments for both M-tiles (two ds_load_b128 each)
      v16bf am[2];
#pragma unroll
      for (int m = 0; m < 2; ++m) {
        const u16* ap = &A[m * 16 + ll][kc * 32 + 8 * lh];
        v8u lo = *(const v8u*)ap;         // K: kc*32 +  8*lh + 0..7
        v8u hi = *(const v8u*)(ap + 16);  // K: kc*32 + 16 + 8*lh + 0..7
        v16u au = __builtin_shufflevector(lo, hi, 0, 1, 2, 3, 4, 5, 6, 7,
                                          8, 9, 10, 11, 12, 13, 14, 15);
        am[m] = __builtin_bit_cast(v16bf, au);
      }
      // B fragments: one 32B contiguous load per lane (L2-resident weights)
#pragma unroll
      for (int g = 0; g < 4; ++g) {
#pragma unroll
        for (int j = 0; j < 2; ++j) {
          int nt = g * 16 + w * 2 + j;
          v16u bu = *(const v16u*)(wp + ((nt * NKC + kc) << 9) + (lane << 4));
          v16bf bb = __builtin_bit_cast(v16bf, bu);
#pragma unroll
          for (int m = 0; m < 2; ++m) {
            acc[m][g][j] = __builtin_amdgcn_wmma_f32_16x16x32_bf16(
                false, am[m], false, bb, (short)0, acc[m][g][j], false, false);
          }
        }
      }
    }
    __syncthreads();   // all WMMA reads of h_t done before h_{t+1} rewrite

    // Gate math + state update; write h to output and back into LDS (bf16)
#pragma unroll
    for (int m = 0; m < 2; ++m) {
#pragma unroll
      for (int j = 0; j < 2; ++j) {
#pragma unroll
        for (int r = 0; r < 8; ++r) {
          float zi = acc[m][0][j][r];
          float zf = acc[m][1][j][r];
          float zg = acc[m][2][j][r];
          float zo = acc[m][3][j][r];
          float ig = sigm(zi);
          float fg = sigm(zf);
          float gg = fast_tanh(zg);
          float og = sigm(zo);
          float cn = fg * cst[m][j][r] + ig * gg;
          cst[m][j][r] = cn;
          float hn = og * fast_tanh(cn);
          int rl = m * 16 + r + 8 * lh;              // local row
          int u  = w * 32 + j * 16 + ll;             // unit
          out[(row0 + rl) * (LSTM_T * LSTM_U) + t * LSTM_U + u] = hn;
          A[rl][LSTM_F + u] = f2bf(hn);              // h_{t+1} input
        }
      }
    }
  }
}

// ---------------------------------------------------------------------------
extern "C" void kernel_launch(void* const* d_in, const int* in_sizes, int n_in,
                              void* d_out, int out_size, void* d_ws,
                              size_t ws_size, hipStream_t stream) {
  const float* x  = (const float*)d_in[0];  // [B,T,F]
  const float* h0 = (const float*)d_in[1];  // [B,U]
  const float* c0 = (const float*)d_in[2];  // [B,U]
  const float* Wk = (const float*)d_in[3];  // [F,4U]
  const float* Wr = (const float*)d_in[4];  // [U,4U]
  const float* b  = (const float*)d_in[5];  // [4U]
  u16* wsB = (u16*)d_ws;                    // 64*10*512 bf16 = 640 KB

  const int packN = NNT * NKC * 512;
  lstm_pack_weights<<<(packN + 255) / 256, 256, 0, stream>>>(Wk, Wr, wsB);

  const int nBlocks = 32768 / BM;           // 1024
  lstm_main<<<nBlocks, 256, 0, stream>>>(x, h0, c0, b, wsB, (float*)d_out);
}
